// Decoder_4088808866248
// MI455X (gfx1250) — matrix operand order repaired and checked by reading, not compiled, not run on hardware
//
#include <hip/hip_runtime.h>
#include <hip/hip_bf16.h>

// ---------------- types ----------------
typedef __attribute__((ext_vector_type(16))) __bf16 v16bf;
typedef __attribute__((ext_vector_type(8)))  float  v8f;
typedef __attribute__((ext_vector_type(4)))  unsigned int u32x4;
typedef __attribute__((ext_vector_type(4)))  float  f32x4;

// Problem constants
#define VSZ 32000
#define EDIM 512
#define EHDIM 1024
#define HDIM 512
#define BB 16
#define SS 128
#define M_ROWS (BB * SS)      // 2048
#define G3H (3 * HDIM)        // 1536

// ---------------- helpers ----------------
__device__ inline unsigned short f32_to_bf16_bits(float f) {
    unsigned int u = __float_as_uint(f);
    // round-to-nearest-even
    unsigned int r = u + 0x7FFFu + ((u >> 16) & 1u);
    return (unsigned short)(r >> 16);
}

union FragU {
    u32x4 q[2];
    v16bf v;
};

// A fragment: 16-bit A matrix 16x32 layout.
// lane r = lane&15 (M row), h = lane>>4.
// h==0: K = [kk+0..7], [kk+16..23] ; h==1: K = [kk+8..15], [kk+24..31]
__device__ inline v16bf load_a_frag_g(const unsigned short* A, int lda,
                                      int mbase, int kk, int r, int h) {
    FragU f;
    const unsigned short* row = A + (size_t)(mbase + r) * lda;
    f.q[0] = *(const u32x4*)(row + kk + 8 * h);
    f.q[1] = *(const u32x4*)(row + kk + 16 + 8 * h);
    return f.v;
}

__device__ inline v16bf load_a_frag_lds(const unsigned short* Alds, int lda,
                                        int kk, int r, int h) {
    FragU f;
    const unsigned short* row = Alds + (size_t)r * lda;
    f.q[0] = *(const u32x4*)(row + kk + 8 * h);
    f.q[1] = *(const u32x4*)(row + kk + 16 + 8 * h);
    return f.v;
}

// B fragment: B is KxN, supplied as Bt[N,K] row-major (weight rows are N).
// lane r = lane&15 (N col), h = lane>>4: K = [kk+8*h .. +7], [kk+16+8*h .. +7]
__device__ inline v16bf load_b_frag_g(const unsigned short* Bt, int ldb,
                                      int nbase, int kk, int r, int h) {
    FragU f;
    const unsigned short* p = Bt + (size_t)(nbase + r) * ldb + kk + 8 * h;
    f.q[0] = *(const u32x4*)(p);
    f.q[1] = *(const u32x4*)(p + 16);
    return f.v;
}

__device__ inline float sigmoidf_(float x) { return 1.0f / (1.0f + __expf(-x)); }

// ---------------- kernel: h0 = hn @ fc_w.T + fc_b ----------------
__global__ void __launch_bounds__(256)
h0_kernel(const float* __restrict__ hn, const float* __restrict__ fc_w,
          const float* __restrict__ fc_b, float* __restrict__ h0) {
    int idx = blockIdx.x * 256 + threadIdx.x;   // 0 .. 16*512-1
    int b = idx >> 9;
    int j = idx & 511;
    const f32x4* x = (const f32x4*)(hn + (size_t)b * EHDIM);
    const f32x4* w = (const f32x4*)(fc_w + (size_t)j * EHDIM);
    float acc = 0.0f;
    #pragma unroll 4
    for (int k = 0; k < EHDIM / 4; ++k) {
        f32x4 a = x[k], c = w[k];
        acc += a.x * c.x + a.y * c.y + a.z * c.z + a.w * c.w;
    }
    h0[idx] = acc + fc_b[j];
}

// ---------------- kernel: gather embedding rows -> bf16 ----------------
__global__ void __launch_bounds__(256)
gather_embed_kernel(const int* __restrict__ y, const float* __restrict__ emb,
                    unsigned short* __restrict__ xb) {
    int idx = blockIdx.x * 256 + threadIdx.x;   // 0 .. 2048*512-1
    int row = idx >> 9;
    int j = idx & 511;
    int tok = y[row];
    xb[idx] = f32_to_bf16_bits(emb[(size_t)tok * EDIM + j]);
}

// ---------------- kernel: f32 -> bf16 convert ----------------
__global__ void __launch_bounds__(256)
cvt_bf16_kernel(const float* __restrict__ in, unsigned short* __restrict__ out,
                int n) {
    int i = blockIdx.x * 256 + threadIdx.x;
    if (i < n) out[i] = f32_to_bf16_bits(in[i]);
}

// ---------------- kernel: bf16 WMMA GEMM  C[M,N] = A[M,512] x Bt[N,512]^T + bias ----------------
// grid: (N/256, M/64), block: 256 threads (8 waves). Wave (wm,wn) computes 32x64.
__global__ void __launch_bounds__(256)
gemm_bf16_kernel(const unsigned short* __restrict__ A,
                 const unsigned short* __restrict__ Bt,
                 const float* __restrict__ bias,
                 float* __restrict__ C, int N) {
    const int K = 512;
    int wave = threadIdx.x >> 5;
    int lane = threadIdx.x & 31;
    int r = lane & 15;
    int h = lane >> 4;
    int wm = wave >> 2;           // 0..1
    int wn = wave & 3;            // 0..3
    int mBase = blockIdx.y * 64 + wm * 32;
    int nBase = blockIdx.x * 256 + wn * 64;

    v8f c[2][4];
    #pragma unroll
    for (int i = 0; i < 2; ++i)
        #pragma unroll
        for (int t = 0; t < 4; ++t)
            c[i][t] = (v8f){0.f, 0.f, 0.f, 0.f, 0.f, 0.f, 0.f, 0.f};

    for (int kk = 0; kk < K; kk += 32) {
        v16bf a0 = load_a_frag_g(A, K, mBase, kk, r, h);
        v16bf a1 = load_a_frag_g(A, K, mBase + 16, kk, r, h);
        // prefetch next K-chunk of the first B row this lane will touch
        if (kk + 32 < K)
            __builtin_prefetch(Bt + (size_t)(nBase + r) * K + kk + 32, 0, 1);
        #pragma unroll
        for (int t = 0; t < 4; ++t) {
            v16bf b = load_b_frag_g(Bt, K, nBase + t * 16, kk, r, h);
            c[0][t] = __builtin_amdgcn_wmma_f32_16x16x32_bf16(
                false, a0, false, b, (short)0, c[0][t], false, false);
            c[1][t] = __builtin_amdgcn_wmma_f32_16x16x32_bf16(
                false, a1, false, b, (short)0, c[1][t], false, false);
        }
    }

    #pragma unroll
    for (int t = 0; t < 4; ++t) {
        int ncol = nBase + t * 16 + r;
        float bv = bias[ncol];
        #pragma unroll
        for (int i = 0; i < 8; ++i) {
            int m0 = mBase + 8 * h + i;
            C[(size_t)m0 * N + ncol] = c[0][t][i] + bv;
            C[(size_t)(m0 + 16) * N + ncol] = c[1][t][i] + bv;
        }
    }
}

// ---------------- kernel: GRU scan (single persistent workgroup) ----------------
// gx: [2048,1536] f32, Whh: [1536,512] bf16, b_hh: [1536], h0: [16,512] f32
// hout: [2048,512] bf16
__global__ void __launch_bounds__(1024)
gru_scan_kernel(const float* __restrict__ gx,
                const unsigned short* __restrict__ Whh,
                const float* __restrict__ b_hh,
                const float* __restrict__ h0,
                unsigned short* __restrict__ hout) {
    __shared__ unsigned short h_b16[BB * HDIM];   // 16 KB, WMMA A operand
    __shared__ float hw[BB * G3H];                // 96 KB, h @ Whh^T result

    int tid = threadIdx.x;
    int lane = tid & 31;
    int wave = tid >> 5;        // 0..31
    int r = lane & 15;
    int h = lane >> 4;

    // Each thread owns 8 consecutive h elements in registers.
    float hreg[8];
    int ebase = tid * 8;        // 1024 threads * 8 = 8192 = 16*512
    #pragma unroll
    for (int i = 0; i < 8; ++i) {
        float v = h0[ebase + i];
        hreg[i] = v;
        h_b16[ebase + i] = f32_to_bf16_bits(v);
    }
    __syncthreads();

    for (int s = 0; s < SS; ++s) {
        // ---- matmul phase: hw[16,1536] = h @ Whh^T ----
        for (int t = wave; t < G3H / 16; t += 32) {   // 96 tiles / 32 waves
            int nb = t * 16;
            v8f c = (v8f){0.f, 0.f, 0.f, 0.f, 0.f, 0.f, 0.f, 0.f};
            #pragma unroll 4
            for (int kk = 0; kk < HDIM; kk += 32) {
                v16bf a = load_a_frag_lds(h_b16, HDIM, kk, r, h);
                v16bf b = load_b_frag_g(Whh, HDIM, nb, kk, r, h);
                c = __builtin_amdgcn_wmma_f32_16x16x32_bf16(
                    false, a, false, b, (short)0, c, false, false);
            }
            #pragma unroll
            for (int i = 0; i < 8; ++i)
                hw[(i + 8 * h) * G3H + nb + r] = c[i];
        }
        __syncthreads();

        // ---- gate phase ----
        {
            int b = ebase >> 9;          // batch row (all 8 elems same b)
            int j0 = ebase & 511;
            const float* gxrow = gx + (size_t)(b * SS + s) * G3H;
            const float* hwrow = hw + b * G3H;
            unsigned short* orow = hout + (size_t)(b * SS + s) * HDIM;
            #pragma unroll
            for (int i = 0; i < 8; ++i) {
                int j = j0 + i;
                float rg = sigmoidf_(gxrow[j] + hwrow[j] + b_hh[j]);
                float zg = sigmoidf_(gxrow[HDIM + j] + hwrow[HDIM + j] +
                                     b_hh[HDIM + j]);
                float ng = tanhf(gxrow[2 * HDIM + j] +
                                 rg * (hwrow[2 * HDIM + j] + b_hh[2 * HDIM + j]));
                float hn_ = (1.0f - zg) * ng + zg * hreg[i];
                hreg[i] = hn_;
                unsigned short hb = f32_to_bf16_bits(hn_);
                h_b16[ebase + i] = hb;
                orow[j] = hb;
            }
        }
        __syncthreads();
    }
}

// ---------------- launch ----------------
extern "C" void kernel_launch(void* const* d_in, const int* in_sizes, int n_in,
                              void* d_out, int out_size, void* d_ws, size_t ws_size,
                              hipStream_t stream) {
    (void)in_sizes; (void)n_in; (void)out_size; (void)ws_size;
    const int*   y      = (const int*)d_in[0];     // [16,128]
    const float* hn     = (const float*)d_in[1];   // [16,1,1024]
    const float* emb    = (const float*)d_in[2];   // [32000,512]
    const float* W_ih   = (const float*)d_in[3];   // [1536,512]
    const float* W_hh   = (const float*)d_in[4];   // [1536,512]
    const float* b_ih   = (const float*)d_in[5];   // [1536]
    const float* b_hh   = (const float*)d_in[6];   // [1536]
    const float* fc_w   = (const float*)d_in[7];   // [512,1024]
    const float* fc_b   = (const float*)d_in[8];   // [512]
    const float* pred_w = (const float*)d_in[9];   // [32000,512]
    const float* pred_b = (const float*)d_in[10];  // [32000]
    float* out = (float*)d_out;                    // [2048,32000]

    // workspace partition (256B aligned chunks)
    char* ws = (char*)d_ws;
    size_t off = 0;
    auto alloc = [&](size_t bytes) {
        char* p = ws + off;
        off += (bytes + 255) & ~(size_t)255;
        return p;
    };
    unsigned short* x_b16    = (unsigned short*)alloc((size_t)M_ROWS * EDIM * 2);
    unsigned short* Wih_b16  = (unsigned short*)alloc((size_t)G3H * EDIM * 2);
    unsigned short* Whh_b16  = (unsigned short*)alloc((size_t)G3H * HDIM * 2);
    unsigned short* pw_b16   = (unsigned short*)alloc((size_t)VSZ * HDIM * 2);
    float*          gx       = (float*)alloc((size_t)M_ROWS * G3H * 4);
    float*          h0       = (float*)alloc((size_t)BB * HDIM * 4);
    unsigned short* hout_b16 = (unsigned short*)alloc((size_t)M_ROWS * HDIM * 2);

    // 1. h0 = hn @ fc_w.T + fc_b
    h0_kernel<<<(BB * HDIM) / 256, 256, 0, stream>>>(hn, fc_w, fc_b, h0);

    // 2. gather + convert embedding rows
    gather_embed_kernel<<<(M_ROWS * EDIM) / 256, 256, 0, stream>>>(y, emb, x_b16);

    // 3. convert weights to bf16
    {
        int n1 = G3H * EDIM;
        cvt_bf16_kernel<<<(n1 + 255) / 256, 256, 0, stream>>>(W_ih, Wih_b16, n1);
        int n2 = G3H * HDIM;
        cvt_bf16_kernel<<<(n2 + 255) / 256, 256, 0, stream>>>(W_hh, Whh_b16, n2);
        int n3 = VSZ * HDIM;
        cvt_bf16_kernel<<<(n3 + 255) / 256, 256, 0, stream>>>(pred_w, pw_b16, n3);
    }

    // 4. gx[2048,1536] = x @ W_ih^T + b_ih   (WMMA bf16)
    {
        dim3 grid(G3H / 256, M_ROWS / 64);
        gemm_bf16_kernel<<<grid, 256, 0, stream>>>(x_b16, Wih_b16, b_ih, gx, G3H);
    }

    // 5. GRU scan (sequential over S) -> hout bf16
    gru_scan_kernel<<<1, 1024, 0, stream>>>(gx, Whh_b16, b_hh, h0, hout_b16);

    // 6. out[2048,32000] = hout @ pred_w^T + pred_b   (WMMA bf16, dominant GEMM)
    {
        dim3 grid(VSZ / 256, M_ROWS / 64);
        gemm_bf16_kernel<<<grid, 256, 0, stream>>>(hout_b16, pw_b16, pred_b, out, VSZ);
    }
}
